// MyLSTMCell_79645873537420
// MI455X (gfx1250) — matrix operand order repaired and checked by reading, not compiled, not run on hardware
//
#include <hip/hip_runtime.h>
#include <hip/hip_bf16.h>
#include <math.h>

// ---------------------------------------------------------------------------
// Fused LSTM cell for MI455X (gfx1250, wave32, WMMA).
//   gates = x @ W + h_prev @ U + b   (W,U per-gate [1024,1024], 4 gates)
//   c = sigmoid(gf)*c_prev + sigmoid(gi)*tanh(gc);  h = sigmoid(go)*tanh(c)
// bf16 WMMA GEMM, f32 accumulation, register-prefetch + double-buffered LDS,
// two-phase K loop with 32-bit strength-reduced global indexing.
// ---------------------------------------------------------------------------

typedef __attribute__((ext_vector_type(16))) __bf16 v16bf;
typedef __attribute__((ext_vector_type(8)))  __bf16 v8bf;
typedef __attribute__((ext_vector_type(8)))  float  v8f;

#define B_ROWS 8192
#define IN_DIM 1024
#define H_DIM  1024

#define M_TILE 128   // rows of batch per workgroup
#define N_TILE 32    // cols of H per workgroup (per gate; 4 gates computed)
#define K_STEP 32    // K per WMMA
#define KPAD   40    // padded K stride in LDS (bf16 elems): 80B rows, 16B aligned

__device__ __forceinline__ float sigmoid_f(float x) {
    return 1.0f / (1.0f + __expf(-x));
}

// Branch-free tanh: 1 - 2/(exp(2x)+1). Saturates correctly for large |x|.
__device__ __forceinline__ float tanh_f(float x) {
    const float e = __expf(2.0f * x);
    return 1.0f - 2.0f / (e + 1.0f);
}

__device__ __forceinline__ unsigned pack_bf16(float a, float b) {
    union { __bf16 h[2]; unsigned u; } p;
    p.h[0] = (__bf16)a;
    p.h[1] = (__bf16)b;
    return p.u;
}

// Build a 16-element bf16 fragment from two 16-byte LDS chunks.
__device__ __forceinline__ v16bf frag_from_lds(const __bf16* rowbase, int k0, int k1) {
    const v8bf lo = *(const v8bf*)(rowbase + k0);
    const v8bf hi = *(const v8bf*)(rowbase + k1);
    return __builtin_shufflevector(lo, hi, 0,1,2,3,4,5,6,7,8,9,10,11,12,13,14,15);
}

__global__ __launch_bounds__(256)
void lstm_fused_wmma_kernel(
    const float* __restrict__ x,  const float* __restrict__ h_prev,
    const float* __restrict__ c_prev,
    const float* __restrict__ Wi, const float* __restrict__ Ui, const float* __restrict__ bi,
    const float* __restrict__ Wf, const float* __restrict__ Uf, const float* __restrict__ bfv,
    const float* __restrict__ Wo, const float* __restrict__ Uo, const float* __restrict__ bo,
    const float* __restrict__ Wc, const float* __restrict__ Uc, const float* __restrict__ bc,
    float* __restrict__ out_h, float* __restrict__ out_c)
{
    // Double-buffered staging: 2 * (10KB A + 10KB B) = 40KB of the 320KB pool.
    __shared__ __bf16 Ash[2][M_TILE * KPAD];       // A tile, row-major [m][k]
    __shared__ __bf16 Bsh[2][4 * N_TILE * KPAD];   // B per gate, [n][k] (transposed)

    const int tid   = threadIdx.x;
    const int lane  = tid & 31;
    const int wave  = tid >> 5;
    const int waveM = wave & 3;     // 4 waves along M -> 32 rows each
    const int waveN = wave >> 2;    // 2 waves along N -> 16 cols each

    const int m0 = blockIdx.y * M_TILE;
    const int n0 = blockIdx.x * N_TILE;

    const v8f vzero = {0.f, 0.f, 0.f, 0.f, 0.f, 0.f, 0.f, 0.f};
    v8f acc[4][2];
#pragma unroll
    for (int g = 0; g < 4; ++g) {
        acc[g][0] = vzero;
        acc[g][1] = vzero;
    }

    // ---- Per-thread staging coordinates / 32-bit base indices ----
    const int a_row0 = tid >> 3;                 // 0..31 (rows step by +32 per chunk)
    const int a_c4   = (tid & 7) * 4;            // K offset 0,4,...,28
    const int b_kp   = (tid >> 4) * 2;           // K pair base 0,2,...,30
    const int b_np   = (tid & 15) * 2;           // N pair base 0,2,...,30

    const int aIdx0 = (m0 + a_row0) * IN_DIM + a_c4;   // + i*32*IN_DIM for chunk i
    const int bIdx0 = b_kp * H_DIM + n0 + b_np;        // + H_DIM for second K row

    // ---- Fragment addressing (constant across K loop) ----
    const int arow0 = waveM * 32 + (lane & 15);        // A tile 0 row
    const int akb   = (lane & 16) ? 8 : 0;             // A: K chunks {kb, kb+16}
    const int brow  = waveN * 16 + (lane & 15);        // B row (= N within tile)
    const int bkb   = (lane & 16) ? 8 : 0;             // B: K chunks {kb, kb+16}

    // ---- Register prefetch buffers ----
    float4 aR[4];
    float2 bR0[4], bR1[4];

    auto load_tiles = [&](const float* __restrict__ Ag,
                          const float* __restrict__ W0, const float* __restrict__ W1,
                          const float* __restrict__ W2, const float* __restrict__ W3,
                          int ai, int bIdx) {
        aR[0] = *(const float4*)(Ag + ai);
        aR[1] = *(const float4*)(Ag + ai + 32 * IN_DIM);
        aR[2] = *(const float4*)(Ag + ai + 64 * IN_DIM);
        aR[3] = *(const float4*)(Ag + ai + 96 * IN_DIM);
        bR0[0] = *(const float2*)(W0 + bIdx);
        bR1[0] = *(const float2*)(W0 + bIdx + H_DIM);
        bR0[1] = *(const float2*)(W1 + bIdx);
        bR1[1] = *(const float2*)(W1 + bIdx + H_DIM);
        bR0[2] = *(const float2*)(W2 + bIdx);
        bR1[2] = *(const float2*)(W2 + bIdx + H_DIM);
        bR0[3] = *(const float2*)(W3 + bIdx);
        bR1[3] = *(const float2*)(W3 + bIdx + H_DIM);
    };

    auto store_tiles = [&](int p) {
#pragma unroll
        for (int i = 0; i < 4; ++i) {
            union { __bf16 h[4]; uint2 u; } pa;
            pa.h[0] = (__bf16)aR[i].x; pa.h[1] = (__bf16)aR[i].y;
            pa.h[2] = (__bf16)aR[i].z; pa.h[3] = (__bf16)aR[i].w;
            *(uint2*)(&Ash[p][(a_row0 + i * 32) * KPAD + a_c4]) = pa.u;  // 8B ds_store
        }
#pragma unroll
        for (int g = 0; g < 4; ++g) {
            __bf16* dst = &Bsh[p][g * (N_TILE * KPAD)];
            // Transpose: pack K-adjacent pair per N column -> 4B ds_store
            *(unsigned*)(dst + (b_np + 0) * KPAD + b_kp) = pack_bf16(bR0[g].x, bR1[g].x);
            *(unsigned*)(dst + (b_np + 1) * KPAD + b_kp) = pack_bf16(bR0[g].y, bR1[g].y);
        }
    };

    auto compute_step = [&](int p) {
        v16bf afrag0 = frag_from_lds(&Ash[p][(arow0     ) * KPAD], akb, akb + 16);
        v16bf afrag1 = frag_from_lds(&Ash[p][(arow0 + 16) * KPAD], akb, akb + 16);
#pragma unroll
        for (int g = 0; g < 4; ++g) {
            v16bf bfrag = frag_from_lds(&Bsh[p][g * (N_TILE * KPAD) + brow * KPAD],
                                        bkb, bkb + 16);
            acc[g][0] = __builtin_amdgcn_wmma_f32_16x16x32_bf16(
                false, afrag0, false, bfrag, (short)0, acc[g][0], false, false);
            acc[g][1] = __builtin_amdgcn_wmma_f32_16x16x32_bf16(
                false, afrag1, false, bfrag, (short)0, acc[g][1], false, false);
        }
    };

    // ---- Pipeline over two phases: (x @ W), then (h_prev @ U) ----
    int p = 0;
    load_tiles(x, Wi, Wf, Wo, Wc, aIdx0, bIdx0);

    // Phase 1: x @ W. Strength-reduced indices: ai += 32, bIdx += 32*H.
    for (int kt = 0; kt < 32; ++kt) {
        store_tiles(p);
        __syncthreads();
        if (kt < 31) {
            load_tiles(x, Wi, Wf, Wo, Wc,
                       aIdx0 + (kt + 1) * K_STEP,
                       bIdx0 + (kt + 1) * K_STEP * H_DIM);
        } else {
            // Boundary: prefetch first tile of phase 2 (h_prev @ U, kk = 0)
            load_tiles(h_prev, Ui, Uf, Uo, Uc, aIdx0, bIdx0);
        }
        compute_step(p);
        p ^= 1;
    }

    // Phase 2: h_prev @ U.
    for (int kt = 0; kt < 32; ++kt) {
        store_tiles(p);
        __syncthreads();
        if (kt < 31) {
            load_tiles(h_prev, Ui, Uf, Uo, Uc,
                       aIdx0 + (kt + 1) * K_STEP,
                       bIdx0 + (kt + 1) * K_STEP * H_DIM);
        }
        compute_step(p);
        p ^= 1;
    }

    // ---- Fused LSTM epilogue ----
    const int colg  = n0 + waveN * 16 + (lane & 15);
    const float biasI = bi[colg];
    const float biasF = bfv[colg];
    const float biasO = bo[colg];
    const float biasC = bc[colg];
    const int rhalf = (lane & 16) ? 8 : 0;

#pragma unroll
    for (int mt = 0; mt < 2; ++mt) {
#pragma unroll
        for (int r = 0; r < 8; ++r) {
            const int rowg = m0 + waveM * 32 + mt * 16 + r + rhalf;
            const int off = rowg * H_DIM + colg;          // < 2^23, fits int
            const float gi = sigmoid_f(acc[0][mt][r] + biasI);
            const float gf = sigmoid_f(acc[1][mt][r] + biasF);
            const float go = sigmoid_f(acc[2][mt][r] + biasO);
            const float gc = tanh_f(acc[3][mt][r] + biasC);
            const float cp = c_prev[off];
            const float cn = gf * cp + gi * gc;
            const float hn = go * tanh_f(cn);
            out_h[off] = hn;
            out_c[off] = cn;
        }
    }
}

extern "C" void kernel_launch(void* const* d_in, const int* in_sizes, int n_in,
                              void* d_out, int out_size, void* d_ws, size_t ws_size,
                              hipStream_t stream) {
    (void)in_sizes; (void)n_in; (void)out_size; (void)d_ws; (void)ws_size;

    const float* x      = (const float*)d_in[0];
    const float* h_prev = (const float*)d_in[1];
    const float* c_prev = (const float*)d_in[2];
    const float* Wi = (const float*)d_in[3];
    const float* Ui = (const float*)d_in[4];
    const float* bi = (const float*)d_in[5];
    const float* Wf = (const float*)d_in[6];
    const float* Uf = (const float*)d_in[7];
    const float* bf = (const float*)d_in[8];
    const float* Wo = (const float*)d_in[9];
    const float* Uo = (const float*)d_in[10];
    const float* bo = (const float*)d_in[11];
    const float* Wc = (const float*)d_in[12];
    const float* Uc = (const float*)d_in[13];
    const float* bc = (const float*)d_in[14];

    float* out_h = (float*)d_out;
    float* out_c = out_h + (size_t)B_ROWS * H_DIM;

    dim3 grid(H_DIM / N_TILE, B_ROWS / M_TILE);   // (32, 64)
    dim3 block(256);
    lstm_fused_wmma_kernel<<<grid, block, 0, stream>>>(
        x, h_prev, c_prev,
        Wi, Ui, bi, Wf, Uf, bf, Wo, Uo, bo, Wc, Uc, bc,
        out_h, out_c);
}
